// GptOssExperts_46926812676210
// MI455X (gfx1250) — compile-verified, matrix operand order checked by reading, not run on hardware
//
#include <hip/hip_runtime.h>
#include <hip/hip_bf16.h>
#include <math.h>

// Problem constants (match reference)
#define NE    8
#define TOK   2048          // B*S
#define HD    2880          // H
#define DD    2880          // D
#define GUP   5760          // 2*D
#define ALPHA 1.702f
#define LIMIT 7.0f

// Tile config: block = 256 threads = 8 waves (wave32)
// Block tile: 256 (M) x 96 (N), K staged 32 per LDS buffer, double-buffered.
// Wave grid 4(M) x 2(N): wave tile 64 x 48 => 4 x 3 WMMA 16x16 subtiles.
#define KTILE 32
#define LDS_S 40            // LDS row stride in ushorts (80B = 5*16B)

typedef __attribute__((ext_vector_type(16))) __bf16          v16bf;
typedef __attribute__((ext_vector_type(8)))  float           v8f;
typedef __attribute__((ext_vector_type(8)))  unsigned short  v8u;
typedef __attribute__((ext_vector_type(4)))  int             v4i;

#define AS1 __attribute__((address_space(1)))
#define AS3 __attribute__((address_space(3)))

#if defined(__has_builtin)
#  if __has_builtin(__builtin_amdgcn_global_load_async_to_lds_b128)
#    define ASYNC_A 1
#  endif
#endif
#ifndef ASYNC_A
#  define ASYNC_A 0
#endif

__device__ __forceinline__ void wait_async() {
#if defined(__has_builtin) && __has_builtin(__builtin_amdgcn_s_wait_asynccnt)
    __builtin_amdgcn_s_wait_asynccnt(0);
#else
    asm volatile("s_wait_asynccnt 0x0" ::: "memory");
#endif
}

__device__ __forceinline__ unsigned short f2bf(float f) {
    unsigned u = __builtin_bit_cast(unsigned, f);
    u += 0x7FFFu + ((u >> 16) & 1u);          // round-to-nearest-even
    return (unsigned short)(u >> 16);
}

// One 16x16x32 bf16 operand fragment from an LDS row-major (K-contig) tile.
// Per ISA 7.12.2: elems 0..7 = K(half*8+i), elems 8..15 = K(16+half*8+i).
__device__ __forceinline__ v16bf ldfrag(const unsigned short* base, int koff) {
    union { v8u p[2]; v16bf v; } u;
    u.p[0] = *(const v8u*)(base + koff);
    u.p[1] = *(const v8u*)(base + koff + 16);
    return u.v;
}

// ---------------- Kernel 0: cast activations f32 -> bf16 ----------------
__global__ __launch_bounds__(256) void cast_x_bf16(const float* __restrict__ x,
                                                   unsigned short* __restrict__ xb) {
    size_t i = ((size_t)blockIdx.x * 256 + threadIdx.x) * 4;   // exact
    float4 f = *(const float4*)(x + i);
    union { unsigned short s[4]; uint2 u; } o;
    o.s[0] = f2bf(f.x); o.s[1] = f2bf(f.y); o.s[2] = f2bf(f.z); o.s[3] = f2bf(f.w);
    *(uint2*)(xb + i) = o.u;
}

// ---------------- Kernel 1: gate_up GEMM + clamped SwiGLU * routing -------
// grid: (GUP/96, TOK/256, NE)
__global__ __launch_bounds__(256) void gemm_gate_up(
    const unsigned short* __restrict__ xb,      // [TOK][HD] bf16
    const float* __restrict__ w,                // [NE][HD][GUP] f32
    const float* __restrict__ bias,             // [NE][GUP]
    const float* __restrict__ routing,          // [TOK][NE]
    unsigned short* __restrict__ hid)           // [NE][TOK][DD] bf16 (routing-scaled)
{
    __shared__ __align__(16) unsigned short lA[2][256 * LDS_S];
    __shared__ __align__(16) unsigned short lB[2][96 * LDS_S];

    const int tid = threadIdx.x;
    const int lane = tid & 31, wid = tid >> 5;
    const int wm = wid >> 1, wn = wid & 1;
    const int row16 = lane & 15, half = lane >> 4;

    const int e  = blockIdx.z;
    const int t0 = blockIdx.y * 256;
    const int n0 = blockIdx.x * 96;
    const float* wbase = w + (size_t)e * HD * GUP;

    const v8f zf = {0.f,0.f,0.f,0.f,0.f,0.f,0.f,0.f};
    v8f acc[4][3];
#pragma unroll
    for (int i = 0; i < 4; ++i)
#pragma unroll
        for (int j = 0; j < 3; ++j) acc[i][j] = zf;

    float4 Breg[3];
    v8u    Areg[4];

    // A tile: 256 rows x 32 k bf16 -> 1024 16B chunks, 4 per thread
    auto issueA = [&](int bufi, int k0) {
#pragma unroll
        for (int i = 0; i < 4; ++i) {
            int c = tid + i * 256, row = c >> 2, kc = (c & 3) * 8;
            const unsigned short* gp = xb + (size_t)(t0 + row) * HD + k0 + kc;
#if ASYNC_A
            __builtin_amdgcn_global_load_async_to_lds_b128(
                (AS1 v4i*)gp, (AS3 v4i*)&lA[bufi][row * LDS_S + kc], 0, 0);
#else
            (void)bufi; Areg[i] = *(const v8u*)gp;
#endif
        }
    };
    auto storeA = [&](int bufi) {
#if !ASYNC_A
#pragma unroll
        for (int i = 0; i < 4; ++i) {
            int c = tid + i * 256, row = c >> 2, kc = (c & 3) * 8;
            *(v8u*)&lA[bufi][row * LDS_S + kc] = Areg[i];
        }
#else
        (void)bufi;
#endif
    };
    // B tile: 32 k-rows x 96 n f32 -> 768 float4 chunks, 3 per thread
    auto loadB = [&](int k0) {
#pragma unroll
        for (int i = 0; i < 3; ++i) {
            int c = tid + i * 256, k = c / 24, nc = (c % 24) * 4;
            Breg[i] = *(const float4*)(wbase + (size_t)(k0 + k) * GUP + n0 + nc);
        }
    };
    auto storeB = [&](int bufi) {
#pragma unroll
        for (int i = 0; i < 3; ++i) {
            int c = tid + i * 256, k = c / 24, nc = (c % 24) * 4;
            lB[bufi][(nc + 0) * LDS_S + k] = f2bf(Breg[i].x);
            lB[bufi][(nc + 1) * LDS_S + k] = f2bf(Breg[i].y);
            lB[bufi][(nc + 2) * LDS_S + k] = f2bf(Breg[i].z);
            lB[bufi][(nc + 3) * LDS_S + k] = f2bf(Breg[i].w);
        }
    };
    auto compute = [&](int bufi) {
        const int koff = half * 8;
        v16bf a[4], b[3];
#pragma unroll
        for (int i = 0; i < 4; ++i)
            a[i] = ldfrag(&lA[bufi][(wm * 64 + i * 16 + row16) * LDS_S], koff);
#pragma unroll
        for (int j = 0; j < 3; ++j)
            b[j] = ldfrag(&lB[bufi][(wn * 48 + j * 16 + row16) * LDS_S], koff);
#pragma unroll
        for (int i = 0; i < 4; ++i)
#pragma unroll
            for (int j = 0; j < 3; ++j)
                acc[i][j] = __builtin_amdgcn_wmma_f32_16x16x32_bf16(
                    false, a[i], false, b[j], (short)0, acc[i][j], false, false);
    };

    // software pipeline: double-buffered LDS, one barrier per K stage
    const int nK = HD / KTILE;
    issueA(0, 0); loadB(0);
    storeB(0); storeA(0);
#if ASYNC_A
    wait_async();
#endif
    __syncthreads();
    int buf = 0;
    for (int kk = 0; kk < nK; ++kk) {
        const bool more = (kk + 1) < nK;
        if (more) { issueA(buf ^ 1, (kk + 1) * KTILE); loadB((kk + 1) * KTILE); }
        compute(buf);
        if (more) {
            storeB(buf ^ 1); storeA(buf ^ 1);
#if ASYNC_A
            wait_async();
#endif
        }
        __syncthreads();
        buf ^= 1;
    }

    // Epilogue: +bias, pair (gate,up) across adjacent lanes, clamped SwiGLU,
    // scale by routing[t,e] so the down GEMM can sum experts in one pass.
    const float* bb = bias + (size_t)e * GUP;
    unsigned short* hbase = hid + (size_t)e * TOK * DD;
#pragma unroll
    for (int i = 0; i < 4; ++i) {
        const int mbase = t0 + wm * 64 + i * 16 + half * 8;
#pragma unroll
        for (int j = 0; j < 3; ++j) {
            const int col = n0 + wn * 48 + j * 16 + row16;
            const float bv = bb[col];
#pragma unroll
            for (int r = 0; r < 8; ++r) {
                float v = acc[i][j][r] + bv;
                float o = __shfl_xor(v, 1, 32);        // partner column
                if ((lane & 1) == 0) {                 // even col = gate, odd = up
                    float g = fminf(v, LIMIT);
                    float u = fminf(fmaxf(o, -LIMIT), LIMIT);
                    float glu = g / (1.f + __expf(-ALPHA * g));
                    float rwv = routing[(size_t)(mbase + r) * NE + e];
                    hbase[(size_t)(mbase + r) * DD + (col >> 1)] =
                        f2bf((u + 1.f) * glu * rwv);
                }
            }
        }
    }
}

// ---------------- Kernel 2: down GEMM, K = NE*DD merged across experts ----
// grid: (HD/96, TOK/256)
__global__ __launch_bounds__(256) void gemm_down(
    const unsigned short* __restrict__ hid,     // [NE][TOK][DD] bf16 (routing-scaled)
    const float* __restrict__ w,                // [NE][DD][HD] f32
    const float* __restrict__ bias,             // [NE][HD]
    const float* __restrict__ routing,          // [TOK][NE]
    float* __restrict__ out)                    // [TOK][HD]
{
    __shared__ __align__(16) unsigned short lA[2][256 * LDS_S];
    __shared__ __align__(16) unsigned short lB[2][96 * LDS_S];

    const int tid = threadIdx.x;
    const int lane = tid & 31, wid = tid >> 5;
    const int wm = wid >> 1, wn = wid & 1;
    const int row16 = lane & 15, half = lane >> 4;

    const int t0 = blockIdx.y * 256;
    const int h0 = blockIdx.x * 96;

    const v8f zf = {0.f,0.f,0.f,0.f,0.f,0.f,0.f,0.f};
    v8f acc[4][3];
#pragma unroll
    for (int i = 0; i < 4; ++i)
#pragma unroll
        for (int j = 0; j < 3; ++j) acc[i][j] = zf;

    float4 Breg[3];
    v8u    Areg[4];

    auto issueA = [&](int bufi, int k0) {
        const int e = k0 / DD, kin = k0 - e * DD;       // KTILE never crosses experts
        const unsigned short* abase = hid + ((size_t)e * TOK + t0) * DD + kin;
#pragma unroll
        for (int i = 0; i < 4; ++i) {
            int c = tid + i * 256, row = c >> 2, kc = (c & 3) * 8;
            const unsigned short* gp = abase + (size_t)row * DD + kc;
#if ASYNC_A
            __builtin_amdgcn_global_load_async_to_lds_b128(
                (AS1 v4i*)gp, (AS3 v4i*)&lA[bufi][row * LDS_S + kc], 0, 0);
#else
            (void)bufi; Areg[i] = *(const v8u*)gp;
#endif
        }
    };
    auto storeA = [&](int bufi) {
#if !ASYNC_A
#pragma unroll
        for (int i = 0; i < 4; ++i) {
            int c = tid + i * 256, row = c >> 2, kc = (c & 3) * 8;
            *(v8u*)&lA[bufi][row * LDS_S + kc] = Areg[i];
        }
#else
        (void)bufi;
#endif
    };
    auto loadB = [&](int k0) {
        const int e = k0 / DD, kin = k0 - e * DD;
        const float* wbase = w + (size_t)e * DD * HD + (size_t)kin * HD + h0;
#pragma unroll
        for (int i = 0; i < 3; ++i) {
            int c = tid + i * 256, k = c / 24, nc = (c % 24) * 4;
            Breg[i] = *(const float4*)(wbase + (size_t)k * HD + nc);
        }
    };
    auto storeB = [&](int bufi) {
#pragma unroll
        for (int i = 0; i < 3; ++i) {
            int c = tid + i * 256, k = c / 24, nc = (c % 24) * 4;
            lB[bufi][(nc + 0) * LDS_S + k] = f2bf(Breg[i].x);
            lB[bufi][(nc + 1) * LDS_S + k] = f2bf(Breg[i].y);
            lB[bufi][(nc + 2) * LDS_S + k] = f2bf(Breg[i].z);
            lB[bufi][(nc + 3) * LDS_S + k] = f2bf(Breg[i].w);
        }
    };
    auto compute = [&](int bufi) {
        const int koff = half * 8;
        v16bf a[4], b[3];
#pragma unroll
        for (int i = 0; i < 4; ++i)
            a[i] = ldfrag(&lA[bufi][(wm * 64 + i * 16 + row16) * LDS_S], koff);
#pragma unroll
        for (int j = 0; j < 3; ++j)
            b[j] = ldfrag(&lB[bufi][(wn * 48 + j * 16 + row16) * LDS_S], koff);
#pragma unroll
        for (int i = 0; i < 4; ++i)
#pragma unroll
            for (int j = 0; j < 3; ++j)
                acc[i][j] = __builtin_amdgcn_wmma_f32_16x16x32_bf16(
                    false, a[i], false, b[j], (short)0, acc[i][j], false, false);
    };

    const int nK = (NE * DD) / KTILE;   // 720
    issueA(0, 0); loadB(0);
    storeB(0); storeA(0);
#if ASYNC_A
    wait_async();
#endif
    __syncthreads();
    int buf = 0;
    for (int kk = 0; kk < nK; ++kk) {
        const bool more = (kk + 1) < nK;
        if (more) { issueA(buf ^ 1, (kk + 1) * KTILE); loadB((kk + 1) * KTILE); }
        compute(buf);
        if (more) {
            storeB(buf ^ 1); storeA(buf ^ 1);
#if ASYNC_A
            wait_async();
#endif
        }
        __syncthreads();
        buf ^= 1;
    }

    // Epilogue: out = acc + sum_e routing[t,e] * down_bias[e][h]
    float bcol[3][NE];
#pragma unroll
    for (int j = 0; j < 3; ++j) {
        const int col = h0 + wn * 48 + j * 16 + row16;
#pragma unroll
        for (int e2 = 0; e2 < NE; ++e2) bcol[j][e2] = bias[(size_t)e2 * HD + col];
    }
#pragma unroll
    for (int i = 0; i < 4; ++i) {
        const int mbase = t0 + wm * 64 + i * 16 + half * 8;
#pragma unroll
        for (int r = 0; r < 8; ++r) {
            const float* rp = routing + (size_t)(mbase + r) * NE;
            float4 r0 = *(const float4*)rp, r1 = *(const float4*)(rp + 4);
            const float rw8[NE] = {r0.x, r0.y, r0.z, r0.w, r1.x, r1.y, r1.z, r1.w};
#pragma unroll
            for (int j = 0; j < 3; ++j) {
                const int col = h0 + wn * 48 + j * 16 + row16;
                float s = acc[i][j][r];
#pragma unroll
                for (int e2 = 0; e2 < NE; ++e2) s += rw8[e2] * bcol[j][e2];
                out[(size_t)(mbase + r) * HD + col] = s;
            }
        }
    }
}

extern "C" void kernel_launch(void* const* d_in, const int* in_sizes, int n_in,
                              void* d_out, int out_size, void* d_ws, size_t ws_size,
                              hipStream_t stream) {
    const float* x   = (const float*)d_in[0];   // (B,S,H)
    const float* rw  = (const float*)d_in[1];   // (T,E)
    const float* gup = (const float*)d_in[2];   // (E,H,2D)
    const float* gub = (const float*)d_in[3];   // (E,2D)
    const float* dp  = (const float*)d_in[4];   // (E,D,H)
    const float* dpb = (const float*)d_in[5];   // (E,H)
    float* out = (float*)d_out;

    // Workspace: bf16 activations (T*H) + bf16 hidden (E*T*D)  (~101 MB)
    unsigned short* xb  = (unsigned short*)d_ws;
    unsigned short* hid = xb + (size_t)TOK * HD;

    cast_x_bf16<<<(TOK * HD) / 1024, 256, 0, stream>>>(x, xb);
    gemm_gate_up<<<dim3(GUP / 96, TOK / 256, NE), 256, 0, stream>>>(xb, gup, gub, rw, hid);
    gemm_down<<<dim3(HD / 96, TOK / 256), 256, 0, stream>>>(hid, dp, dpb, rw, out);
}
